// WeightedTopDownTreeLSTMEndtoEnd_32238024524221
// MI455X (gfx1250) — compile-verified, matrix operand order checked by reading, not run on hardware
//
#include <hip/hip_runtime.h>
#include <math.h>

typedef __attribute__((ext_vector_type(16))) __bf16 v16bf;
typedef __attribute__((ext_vector_type(8)))  float  v8f;

#define N_NODES 1111
#define BATCH   128
#define DIN     300
#define MDIM    256
#define KX      320      // Din padded to multiple of 32
#define KTOT    576      // KX + MDIM
#define NCHUNK  18       // KTOT / 32
#define LDS_STRIDE 40    // 32 + 8 pad (bf16 elements)
#define A_BUF   (BATCH * LDS_STRIDE)
#define B_BUF   (128 * LDS_STRIDE)

union FragU { v16bf v; unsigned int u[8]; };

// ---------------------------------------------------------------------------
// Pack [Wioux;Wiouh] and [Wfx;Wfh] into bf16 Wperm[8][128][KTOT], with block
// columns interleaved as (group g in {i,o,u,f}) x (16 j-columns) so each
// wave's 64-wide column tile carries all four gates for the same j's.
// col c (0..127): g = (c>>4)&3 ; jj = ((c>>6)<<4) | (c&15) ; j = jb*32 + jj
// ---------------------------------------------------------------------------
__global__ __launch_bounds__(256) void prep_weights_kernel(
    const float* __restrict__ Wioux, const float* __restrict__ Wiouh,
    const float* __restrict__ Wfx,   const float* __restrict__ Wfh,
    __bf16* __restrict__ wperm)
{
  int idx = blockIdx.x * 256 + threadIdx.x;      // exactly 8*128*576 threads
  int jb  = idx / (128 * KTOT);
  int rem = idx % (128 * KTOT);
  int c   = rem / KTOT;
  int k   = rem % KTOT;
  int g   = (c >> 4) & 3;
  int jj  = ((c >> 6) << 4) | (c & 15);
  int j   = jb * 32 + jj;
  float v = 0.0f;
  if (g < 3) {
    int row = g * 256 + j;
    if (k < DIN)      v = Wioux[row * DIN + k];
    else if (k >= KX) v = Wiouh[row * MDIM + (k - KX)];
  } else {
    if (k < DIN)      v = Wfx[j * DIN + k];
    else if (k >= KX) v = Wfh[j * MDIM + (k - KX)];
  }
  wperm[idx] = (__bf16)v;
}

// ---------------------------------------------------------------------------
// One tree level: block = (node, 32-col block). 8 waves, wave tile 32x64.
// Fused GEMM over K = 320(x, bf16-converted on the fly) + 256(scaled parent
// h), double-buffered LDS, then per-lane LSTM gating on the accumulators.
// ---------------------------------------------------------------------------
__global__ __launch_bounds__(256) void tree_level_kernel(
    const float* __restrict__ inputs,   // [N, B, DIN]
    const float* __restrict__ prob,     // [N, N]
    const float* __restrict__ bioux, const float* __restrict__ biouh,
    const float* __restrict__ bfx,   const float* __restrict__ bfh,
    const int*   __restrict__ parents,
    const __bf16* __restrict__ wperm,   // [8][128][KTOT]
    float* __restrict__ c_out, float* __restrict__ h_out,  // [N, B, M]
    int node_start, int has_parent)
{
  __shared__ __align__(16) __bf16 lds_a[2 * A_BUF];
  __shared__ __align__(16) __bf16 lds_b[2 * B_BUF];

  const int node = node_start + blockIdx.x;
  const int jb   = blockIdx.y;          // 0..7 -> j range [32*jb, 32*jb+32)
  const int tid  = threadIdx.x;
  const int lane = tid & 31;
  const int wid  = tid >> 5;
  const int wr   = wid & 3;             // wave row (0..3), 32 rows each
  const int wc   = wid >> 2;            // wave col (0..1), 64 cols each

  int   par   = 0;
  float scale = 0.0f;
  if (has_parent) {
    par   = parents[node];
    scale = prob[par * N_NODES + node];
  }

  v8f acc[2][4] = {};                   // [row subtile][gate group]

  // LDS staging assignment: thread t -> row (t>>1), 16 K's at ((t&1)*16)
  const int arow = tid >> 1;
  const int akb  = (tid & 1) * 16;
  const float* xrow = inputs + (size_t)(node * BATCH + arow) * DIN;
  const float* hrow = has_parent ? (h_out + (size_t)(par * BATCH + arow) * MDIM)
                                 : (const float*)0;
  const __bf16* wsrc = wperm + (size_t)(jb * 128 + arow) * KTOT + akb;
  __bf16* dstA = lds_a + arow * LDS_STRIDE + akb;
  __bf16* dstB = lds_b + arow * LDS_STRIDE + akb;

  // Fragment gather constants (documented 16-bit A/B layout, packed pairs)
  const int fm = lane & 15;
  const int ko = (lane < 16) ? 0 : 4;   // uint (bf16-pair) offset in row

  // ---- stage one K-chunk (32 wide) into LDS buffer `buf` ----
  auto stage = [&](int ch, int buf) {
    const int c0 = ch * 32;
    __bf16* da = dstA + buf * A_BUF;
    __bf16* db = dstB + buf * B_BUF;
    // B tile: 16 bf16 = 32 bytes, already packed -> two b128 copies
    {
      const uint4* s4 = (const uint4*)(wsrc + c0);
      ((uint4*)db)[0] = s4[0];
      ((uint4*)db)[1] = s4[1];
    }
    // A tile
    if (c0 + 32 <= DIN) {                           // pure x region
      const float4* s4 = (const float4*)(xrow + c0 + akb);
      #pragma unroll
      for (int q = 0; q < 4; ++q) {
        float4 f = s4[q];
        da[q * 4 + 0] = (__bf16)f.x;
        da[q * 4 + 1] = (__bf16)f.y;
        da[q * 4 + 2] = (__bf16)f.z;
        da[q * 4 + 3] = (__bf16)f.w;
      }
    } else if (c0 >= KX) {                          // recurrent region
      if (has_parent) {
        const float4* s4 = (const float4*)(hrow + (c0 - KX) + akb);
        #pragma unroll
        for (int q = 0; q < 4; ++q) {
          float4 f = s4[q];
          da[q * 4 + 0] = (__bf16)(scale * f.x);
          da[q * 4 + 1] = (__bf16)(scale * f.y);
          da[q * 4 + 2] = (__bf16)(scale * f.z);
          da[q * 4 + 3] = (__bf16)(scale * f.w);
        }
      } else {
        uint4 z = {0u, 0u, 0u, 0u};
        ((uint4*)da)[0] = z;
        ((uint4*)da)[1] = z;
      }
    } else {                                        // boundary chunk (288..320)
      // Unconditional loads with clamped index; select 0 past Din. No exec
      // divergence, no per-load waits.
      #pragma unroll
      for (int kk = 0; kk < 16; ++kk) {
        int k  = c0 + akb + kk;
        int kc = (k < DIN) ? k : 0;
        float v = xrow[kc];
        da[kk] = (__bf16)((k < DIN) ? v : 0.0f);
      }
    }
  };

  stage(0, 0);
  __syncthreads();

  for (int ch = 0; ch < NCHUNK; ++ch) {
    const int cur = ch & 1;
    // Prefetch next chunk into the other buffer (overlaps with WMMAs below).
    if (ch + 1 < NCHUNK) stage(ch + 1, cur ^ 1);

    const __bf16* baseA = lds_a + cur * A_BUF;
    const __bf16* baseB = lds_b + cur * B_BUF;

    FragU fb[4];
    #pragma unroll
    for (int g = 0; g < 4; ++g) {
      const unsigned int* colp =
          (const unsigned int*)(baseB + (wc * 64 + g * 16 + fm) * LDS_STRIDE);
      #pragma unroll
      for (int v = 0; v < 4; ++v) {
        fb[g].u[v]     = colp[ko + v];
        fb[g].u[4 + v] = colp[8 + ko + v];
      }
    }
    #pragma unroll
    for (int sr = 0; sr < 2; ++sr) {
      FragU fa;
      const unsigned int* rowp =
          (const unsigned int*)(baseA + (wr * 32 + sr * 16 + fm) * LDS_STRIDE);
      #pragma unroll
      for (int v = 0; v < 4; ++v) {
        fa.u[v]     = rowp[ko + v];
        fa.u[4 + v] = rowp[8 + ko + v];
      }
      #pragma unroll
      for (int g = 0; g < 4; ++g) {
        acc[sr][g] = __builtin_amdgcn_wmma_f32_16x16x32_bf16(
            false, fa.v, false, fb[g].v, (short)0, acc[sr][g], false, false);
      }
    }
    __syncthreads();
  }

  // ---- per-lane LSTM gating on accumulators, store c & h ----
  const int n  = lane & 15;                 // output column within 16
  const int mh = (lane >> 4) << 3;          // +8 rows for upper half-wave
  const int j  = jb * 32 + wc * 16 + n;     // output column 0..255
  const float bi = bioux[j]       + biouh[j];
  const float bo = bioux[256 + j] + biouh[256 + j];
  const float bu = bioux[512 + j] + biouh[512 + j];
  const float bf = bfx[j]         + bfh[j];

  #pragma unroll
  for (int sr = 0; sr < 2; ++sr) {
    #pragma unroll
    for (int r = 0; r < 8; ++r) {
      const int b  = wr * 32 + sr * 16 + mh + r;   // batch row
      float iv = acc[sr][0][r] + bi;
      float ov = acc[sr][1][r] + bo;
      float uv = acc[sr][2][r] + bu;
      float fv = acc[sr][3][r] + bf;
      float pc = has_parent ? scale * c_out[(size_t)(par * BATCH + b) * MDIM + j]
                            : 0.0f;
      float si = 1.0f / (1.0f + expf(-iv));
      float so = 1.0f / (1.0f + expf(-ov));
      float sf = 1.0f / (1.0f + expf(-fv));
      float cn = si * tanhf(uv) + sf * pc;
      float hn = so * tanhf(cn);
      size_t o = (size_t)(node * BATCH + b) * MDIM + j;
      c_out[o] = cn;
      h_out[o] = hn;
    }
  }
}

// ---------------------------------------------------------------------------
extern "C" void kernel_launch(void* const* d_in, const int* in_sizes, int n_in,
                              void* d_out, int out_size, void* d_ws, size_t ws_size,
                              hipStream_t stream) {
  (void)in_sizes; (void)n_in; (void)out_size; (void)ws_size;

  const float* inputs = (const float*)d_in[0];
  const float* prob   = (const float*)d_in[1];
  const float* Wioux  = (const float*)d_in[2];
  const float* bioux  = (const float*)d_in[3];
  const float* Wiouh  = (const float*)d_in[4];
  const float* biouh  = (const float*)d_in[5];
  const float* Wfx    = (const float*)d_in[6];
  const float* bfx    = (const float*)d_in[7];
  const float* Wfh    = (const float*)d_in[8];
  const float* bfh    = (const float*)d_in[9];
  const int*   parents = (const int*)d_in[10];

  float* c_out = (float*)d_out;                                  // [N,B,M]
  float* h_out = c_out + (size_t)N_NODES * BATCH * MDIM;         // [N,B,M]
  __bf16* wperm = (__bf16*)d_ws;                                 // ~1.2 MB

  // Pack weights to bf16 in the WMMA-friendly interleaved layout.
  prep_weights_kernel<<<(8 * 128 * KTOT) / 256, 256, 0, stream>>>(
      Wioux, Wiouh, Wfx, Wfh, wperm);

  // Level-synchronous sweep; levels are contiguous node ranges for a 10-ary
  // heap-style tree. Stream ordering provides the level dependency.
  const int starts[4] = {0, 1, 11, 111};
  const int counts[4] = {1, 10, 100, 1000};
  for (int d = 0; d < 4; ++d) {
    dim3 grid(counts[d], 8);
    tree_level_kernel<<<grid, 256, 0, stream>>>(
        inputs, prob, bioux, biouh, bfx, bfh, parents, wperm,
        c_out, h_out, starts[d], d > 0 ? 1 : 0);
  }
}
